// AdaCoFNet_74835510166120
// MI455X (gfx1250) — matrix operand order repaired and checked by reading, not compiled, not run on hardware
//
#include <hip/hip_runtime.h>
#include <hip/hip_bf16.h>
#include <cstdint>
#include <cstddef>

// ---------------------------------------------------------------------------
// AdaCoF-style net on gfx1250 (MI455X): implicit-GEMM 3x3 convs through
// v_wmma_f32_16x16x32_bf16 (4 WMMAs / wave / K-step), double-buffered LDS
// (one barrier per K-step), b128-vectorized staging, fused epilogues;
// VALU kernels for softmax and the 49-tap bilinear sampler/blend.
// ---------------------------------------------------------------------------

typedef __bf16 bf16_t;
typedef __attribute__((ext_vector_type(16))) __bf16 v16bf;
typedef __attribute__((ext_vector_type(8)))  __bf16 v8bf;
typedef __attribute__((ext_vector_type(8)))  float  v8f;

#define BB 2
#define HH 256
#define WW 256
#define K2 49

// ---------------------------------------------------------------------------
// Pack: concat(feat1[32], feat2[32], corr[81]) NCHW f32 -> NHWC bf16 padded 160
// ---------------------------------------------------------------------------
__global__ __launch_bounds__(256)
void pack_combined_k(const float* __restrict__ f1, const float* __restrict__ f2,
                     const float* __restrict__ corr, bf16_t* __restrict__ out,
                     int B, int H, int W, int CP) {
    int idx = blockIdx.x * blockDim.x + threadIdx.x;
    int HW = H * W;
    int total = B * CP * HW;
    if (idx >= total) return;
    int x = idx % W;
    int y = (idx / W) % H;
    int c = (idx / HW) % CP;
    int b = idx / (HW * CP);
    float v = 0.0f;
    if (c < 32)        v = f1[((size_t)(b * 32 + c)) * HW + y * W + x];
    else if (c < 64)   v = f2[((size_t)(b * 32 + (c - 32))) * HW + y * W + x];
    else if (c < 145)  v = corr[((size_t)(b * 81 + (c - 64))) * HW + y * W + x];
    out[(((size_t)b * H + y) * W + x) * CP + c] = (bf16_t)v;
}

// ---------------------------------------------------------------------------
// Pack weights: OIHW f32 [Cout][Cin][3][3] -> [CoutP][9*CinP] bf16, zero pad.
// k index = tap*CinP + c  (tap = ky*3+kx)
// ---------------------------------------------------------------------------
__global__ __launch_bounds__(256)
void pack_weights_k(const float* __restrict__ w, bf16_t* __restrict__ wp,
                    int Cin, int CinP, int Cout, int CoutP) {
    int idx = blockIdx.x * blockDim.x + threadIdx.x;
    int total = CoutP * 9 * CinP;
    if (idx >= total) return;
    int c   = idx % CinP;
    int tap = (idx / CinP) % 9;
    int co  = idx / (CinP * 9);
    float v = 0.0f;
    if (co < Cout && c < Cin)
        v = w[((size_t)(co * Cin + c)) * 9 + tap];
    wp[idx] = (bf16_t)v;
}

// ---------------------------------------------------------------------------
// Implicit-GEMM 3x3 SAME conv, bf16 WMMA, f32 accumulate.
//   Block tile: M = 64 output channels x N = 128 pixels (one row segment).
//   8 waves; wave wn owns pixel fragment [wn*16, wn*16+16) and computes all
//   four 16-row M-fragments -> 4 v_wmma_f32_16x16x32_bf16 per K-step.
//   Double-buffered LDS: stage chunk i+1 while computing chunk i; a single
//   barrier per chunk. All LDS traffic is 16B-aligned b128 (fragment runs
//   are K-contiguous in the ISA layouts). No division in the hot loop.
// act: 0 = none, 1 = leaky-relu(0.1), 2 = sigmoid
// ---------------------------------------------------------------------------
__global__ __launch_bounds__(256)
void conv3x3_wmma_k(const bf16_t* __restrict__ inb, int CinP,
                    const bf16_t* __restrict__ wpk, int Ktot,
                    const float* __restrict__ bias,
                    int CoutActual, int CoutP,
                    bf16_t* __restrict__ outb,   // NHWC padded bf16 (or null)
                    float*  __restrict__ outf,   // NCHW f32 (or null)
                    int H, int W, int act) {
    __shared__ bf16_t As[2][64][40];    // [buf][cout][k]  pitch 80B
    __shared__ bf16_t Bs[2][128][40];   // [buf][pixel][k] pitch 80B

    const int lane = threadIdx.x & 31;
    const int wn   = threadIdx.x >> 5;   // wave -> pixel fragment index (0..7)
    const int g    = lane >> 4;          // half-wave group
    const int mn   = lane & 15;

    const int tilesX = W >> 7;           // 128-pixel tiles per row
    int t = blockIdx.x;
    const int xt = t % tilesX;  t /= tilesX;
    const int y  = t % H;       t /= H;
    const int b  = t;
    const int x0 = xt * 128;
    const int coutBase = blockIdx.y * 64;

    // cooperative staging assignments
    const int rowA = threadIdx.x >> 2;         // 0..63
    const int segA = (threadIdx.x & 3) * 8;    // 0,8,16,24 (16B chunks)
    const int pB   = threadIdx.x >> 1;         // 0..127
    const int segB = (threadIdx.x & 1) * 16;   // 0,16 (32B halves)

    v8f acc[4];
#pragma unroll
    for (int mm = 0; mm < 4; ++mm)
#pragma unroll
        for (int i = 0; i < 8; ++i) acc[mm][i] = 0.0f;

    v8bf zero;
#pragma unroll
    for (int i = 0; i < 8; ++i) zero[i] = (bf16_t)0.0f;

    // ---- staging state machine (tap rollover handled outside hot math) ----
    const bf16_t* aSrc = wpk + (size_t)(coutBase + rowA) * Ktot + segA;
    int tap = 0, c0 = 0;
    const bf16_t* bSrc;
    bool okB;
    {
        const int yy = y - 1, xx = x0 + pB - 1;   // tap 0: ky=-1, kx=-1
        okB = (yy >= 0) && (xx >= 0) && (xx < W);
        bSrc = inb + (((size_t)b * H + (okB ? yy : 0)) * W + (okB ? xx : 0)) * CinP + segB;
    }

    auto stage = [&](int buf) {
        // A: one 16B chunk per thread (global b128 -> ds b128)
        *(v8bf*)&As[buf][rowA][segA] = *(const v8bf*)aSrc;
        __builtin_prefetch(aSrc + 32, 0, 3);       // global_prefetch_b8
        aSrc += 32;
        // B: 32B (two b128) per thread, zero-filled at borders
        v8bf v0 = zero, v1 = zero;
        if (okB) {
            v0 = *(const v8bf*)(bSrc);
            v1 = *(const v8bf*)(bSrc + 8);
        }
        *(v8bf*)&Bs[buf][pB][segB]     = v0;
        *(v8bf*)&Bs[buf][pB][segB + 8] = v1;
        bSrc += 32;
        // advance (tap rollover runs 9 times total)
        c0 += 32;
        if (c0 == CinP) {
            c0 = 0;
            ++tap;
            if (tap < 9) {
                const int ky = tap / 3 - 1;
                const int kx = tap % 3 - 1;
                const int yy = y + ky;
                const int xx = x0 + pB + kx;
                okB = (yy >= 0) && (yy < H) && (xx >= 0) && (xx < W);
                bSrc = inb + (((size_t)b * H + (okB ? yy : 0)) * W + (okB ? xx : 0)) * CinP + segB;
            }
        }
    };

    const int nch = Ktot >> 5;   // K chunks of 32
    stage(0);
    __syncthreads();

    for (int chunk = 0; chunk < nch; ++chunk) {
        const int cur = chunk & 1;
        // overlap: stage next chunk into the other buffer
        if (chunk + 1 < nch) stage(cur ^ 1);

        // ---- B fragment: k = g*16 + 0..15, contiguous (2x ds_load_b128)
        v8bf blo = *(const v8bf*)&Bs[cur][wn * 16 + mn][g * 16];
        v8bf bhi = *(const v8bf*)&Bs[cur][wn * 16 + mn][g * 16 + 8];
        v16bf bfr;
#pragma unroll
        for (int i = 0; i < 8; ++i) { bfr[i] = blo[i]; bfr[i + 8] = bhi[i]; }

        // ---- preload all 4 A fragments, then issue 4 independent WMMAs
        v16bf af[4];
#pragma unroll
        for (int mm = 0; mm < 4; ++mm) {
            v8bf alo = *(const v8bf*)&As[cur][mm * 16 + mn][g * 16];
            v8bf ahi = *(const v8bf*)&As[cur][mm * 16 + mn][g * 16 + 8];
#pragma unroll
            for (int i = 0; i < 8; ++i) { af[mm][i] = alo[i]; af[mm][i + 8] = ahi[i]; }
        }
#pragma unroll
        for (int mm = 0; mm < 4; ++mm)
            acc[mm] = __builtin_amdgcn_wmma_f32_16x16x32_bf16(
                false, af[mm], false, bfr, (short)0, acc[mm], false, false);

        __syncthreads();   // single barrier per chunk (ping-pong buffers)
    }

    // ---- epilogue: bias + activation, write bf16 NHWC and/or f32 NCHW
    const int x = x0 + wn * 16 + mn;
#pragma unroll
    for (int mm = 0; mm < 4; ++mm) {
#pragma unroll
        for (int j = 0; j < 8; ++j) {
            const int co = coutBase + mm * 16 + (j + 8 * g);
            if (co < CoutActual) {
                float v = acc[mm][j] + bias[co];
                if (act == 1)      v = (v >= 0.f) ? v : 0.1f * v;
                else if (act == 2) v = 1.0f / (1.0f + __expf(-v));
                if (outb) outb[(((size_t)b * H + y) * W + x) * CoutP + co] = (bf16_t)v;
                if (outf) outf[(((size_t)(b * CoutActual + co)) * H + y) * W + x] = v;
            } else if (outb && co < CoutP) {
                outb[(((size_t)b * H + y) * W + x) * CoutP + co] = (bf16_t)0.0f;
            }
        }
    }
}

// ---------------------------------------------------------------------------
// Channel softmax over 49 logits (NCHW f32 -> NCHW f32)
// ---------------------------------------------------------------------------
__global__ __launch_bounds__(256)
void softmax49_k(const float* __restrict__ logits, float* __restrict__ out,
                 int B, int H, int W) {
    int idx = blockIdx.x * blockDim.x + threadIdx.x;
    int HW = H * W;
    if (idx >= B * HW) return;
    int p = idx % HW;
    int b = idx / HW;
    const float* src = logits + (size_t)b * K2 * HW + p;
    float m = -3.4e38f;
#pragma unroll 7
    for (int k = 0; k < K2; ++k) m = fmaxf(m, src[(size_t)k * HW]);
    float s = 0.f;
#pragma unroll 7
    for (int k = 0; k < K2; ++k) s += __expf(src[(size_t)k * HW] - m);
    float inv = 1.0f / s;
    float* dst = out + (size_t)b * K2 * HW + p;
#pragma unroll 7
    for (int k = 0; k < K2; ++k) dst[(size_t)k * HW] = __expf(src[(size_t)k * HW] - m) * inv;
}

// ---------------------------------------------------------------------------
// 49-tap adaptive bilinear sampling of one frame
// ---------------------------------------------------------------------------
__device__ __forceinline__
void sample_one(const float* __restrict__ img, const float* __restrict__ off,
                const float* __restrict__ wt, int b, int y, int x,
                int H, int W, float acc[3]) {
    const int HW = H * W;
    const float sxs = (float)W / (float)(W - 1);
    const float sys = (float)H / (float)(H - 1);
    const int p = y * W + x;
    for (int k = 0; k < K2; ++k) {
        const float dx = off[(((size_t)b * K2 + k) * 2 + 0) * HW + p];
        const float dy = off[(((size_t)b * K2 + k) * 2 + 1) * HW + p];
        const float wk = wt[((size_t)b * K2 + k) * HW + p];
        float px = fminf(fmaxf(((float)x + dx) * sxs - 0.5f, 0.0f), (float)(W - 1));
        float py = fminf(fmaxf(((float)y + dy) * sys - 0.5f, 0.0f), (float)(H - 1));
        float x0f = floorf(px), y0f = floorf(py);
        float fx = px - x0f, fy = py - y0f;
        int x0i = (int)x0f, y0i = (int)y0f;
        int x1i = min(x0i + 1, W - 1), y1i = min(y0i + 1, H - 1);
        float w00 = (1.f - fx) * (1.f - fy), w01 = fx * (1.f - fy);
        float w10 = (1.f - fx) * fy,         w11 = fx * fy;
#pragma unroll
        for (int c = 0; c < 3; ++c) {
            const float* ic = img + ((size_t)b * 3 + c) * HW;
            float v = ic[y0i * W + x0i] * w00 + ic[y0i * W + x1i] * w01 +
                      ic[y1i * W + x0i] * w10 + ic[y1i * W + x1i] * w11;
            acc[c] += wk * v;
        }
    }
}

__global__ __launch_bounds__(256)
void sample_blend_k(const float* __restrict__ frame1, const float* __restrict__ frame2,
                    const float* __restrict__ off1, const float* __restrict__ off2,
                    const float* __restrict__ wt1,  const float* __restrict__ wt2,
                    const float* __restrict__ blend, float* __restrict__ out,
                    int B, int H, int W) {
    int idx = blockIdx.x * blockDim.x + threadIdx.x;
    int HW = H * W;
    if (idx >= B * HW) return;
    int x = idx % W;
    int y = (idx / W) % H;
    int b = idx / HW;
    float a1[3] = {0.f, 0.f, 0.f};
    float a2[3] = {0.f, 0.f, 0.f};
    sample_one(frame1, off1, wt1, b, y, x, H, W, a1);
    sample_one(frame2, off2, wt2, b, y, x, H, W, a2);
    const float bl = blend[(size_t)b * HW + y * W + x];
#pragma unroll
    for (int c = 0; c < 3; ++c)
        out[((size_t)(b * 3 + c)) * HW + y * W + x] = bl * a1[c] + (1.f - bl) * a2[c];
}

// ---------------------------------------------------------------------------
// Host-side orchestration
// ---------------------------------------------------------------------------
static inline size_t alignUp256(size_t v) { return (v + 255) & ~(size_t)255; }

static void launch_pack_weights(hipStream_t s, const float* w, bf16_t* wp,
                                int Cin, int CinP, int Cout, int CoutP) {
    int total = CoutP * 9 * CinP;
    pack_weights_k<<<(total + 255) / 256, 256, 0, s>>>(w, wp, Cin, CinP, Cout, CoutP);
}

static void launch_conv(hipStream_t s, const bf16_t* inb, int CinP,
                        const bf16_t* wpk, const float* bias,
                        int Cout, int CoutP, bf16_t* outb, float* outf, int act) {
    dim3 grid(BB * HH * (WW / 128), CoutP / 64);
    conv3x3_wmma_k<<<grid, 256, 0, s>>>(inb, CinP, wpk, 9 * CinP, bias,
                                        Cout, CoutP, outb, outf, HH, WW, act);
}

extern "C" void kernel_launch(void* const* d_in, const int* in_sizes, int n_in,
                              void* d_out, int out_size, void* d_ws, size_t ws_size,
                              hipStream_t stream) {
    (void)in_sizes; (void)n_in; (void)out_size;
    const int HW = HH * WW;

    // inputs (setup_inputs dict order)
    const float* frame1 = (const float*)d_in[0];
    const float* frame2 = (const float*)d_in[1];
    const float* feat1  = (const float*)d_in[2];
    const float* feat2  = (const float*)d_in[3];
    const float* corr   = (const float*)d_in[4];
    // k1: 5..14, k2: 15..24, blend: 25..28

    // d_out regions (f32, return order)
    float* out_output = (float*)d_out;
    float* out_blend  = out_output + (size_t)BB * 3 * HW;
    float* out_off1   = out_blend  + (size_t)BB * 1 * HW;
    float* out_off2   = out_off1   + (size_t)BB * 2 * K2 * HW;
    float* out_wt1    = out_off2   + (size_t)BB * 2 * K2 * HW;
    float* out_wt2    = out_wt1    + (size_t)BB * K2 * HW;

    // workspace carve-up
    char* ws = (char*)d_ws;
    size_t o = 0;
    bf16_t* combined = (bf16_t*)(ws + o); o = alignUp256(o + (size_t)BB * HW * 160 * sizeof(bf16_t));
    bf16_t* bufA     = (bf16_t*)(ws + o); o = alignUp256(o + (size_t)BB * HW * 256 * sizeof(bf16_t));
    bf16_t* bufB     = (bf16_t*)(ws + o); o = alignUp256(o + (size_t)BB * HW * 256 * sizeof(bf16_t));
    bf16_t* wpk      = (bf16_t*)(ws + o); o = alignUp256(o + (size_t)256 * 9 * 256 * sizeof(bf16_t));
    float*  logits   = (float*) (ws + o); o = alignUp256(o + (size_t)BB * K2 * HW * sizeof(float));
    if (o > ws_size) return;  // insufficient scratch

    // ---- pack combined input (145 ch -> 160 padded, NHWC bf16)
    {
        int total = BB * 160 * HW;
        pack_combined_k<<<(total + 255) / 256, 256, 0, stream>>>(
            feat1, feat2, corr, combined, BB, HH, WW, 160);
    }

    // ---- two kernel predictors
    for (int p = 0; p < 2; ++p) {
        const int base = 5 + p * 10;
        const float* w1 = (const float*)d_in[base + 0];
        const float* b1 = (const float*)d_in[base + 1];
        const float* w2 = (const float*)d_in[base + 2];
        const float* b2 = (const float*)d_in[base + 3];
        const float* w3 = (const float*)d_in[base + 4];
        const float* b3 = (const float*)d_in[base + 5];
        const float* wo = (const float*)d_in[base + 6];
        const float* bo = (const float*)d_in[base + 7];
        const float* wwt = (const float*)d_in[base + 8];
        const float* bwt = (const float*)d_in[base + 9];
        float* off_dst = (p == 0) ? out_off1 : out_off2;
        float* wt_dst  = (p == 0) ? out_wt1  : out_wt2;

        launch_pack_weights(stream, w1, wpk, 145, 160, 256, 256);
        launch_conv(stream, combined, 160, wpk, b1, 256, 256, bufA, nullptr, 1);

        launch_pack_weights(stream, w2, wpk, 256, 256, 256, 256);
        launch_conv(stream, bufA, 256, wpk, b2, 256, 256, bufB, nullptr, 1);

        launch_pack_weights(stream, w3, wpk, 256, 256, 128, 128);
        launch_conv(stream, bufB, 256, wpk, b3, 128, 128, bufA, nullptr, 1);

        launch_pack_weights(stream, wo, wpk, 128, 128, 2 * K2, 128);
        launch_conv(stream, bufA, 128, wpk, bo, 2 * K2, 128, nullptr, off_dst, 0);

        launch_pack_weights(stream, wwt, wpk, 128, 128, K2, 64);
        launch_conv(stream, bufA, 128, wpk, bwt, K2, 64, nullptr, logits, 0);

        softmax49_k<<<(BB * HW + 255) / 256, 256, 0, stream>>>(
            logits, wt_dst, BB, HH, WW);
    }

    // ---- blend mask branch
    {
        const float* blw1 = (const float*)d_in[25];
        const float* blb1 = (const float*)d_in[26];
        const float* blw2 = (const float*)d_in[27];
        const float* blb2 = (const float*)d_in[28];

        launch_pack_weights(stream, blw1, wpk, 145, 160, 64, 64);
        launch_conv(stream, combined, 160, wpk, blb1, 64, 64, bufB, nullptr, 1);

        launch_pack_weights(stream, blw2, wpk, 64, 64, 1, 64);
        launch_conv(stream, bufB, 64, wpk, blb2, 1, 64, nullptr, out_blend, 2);
    }

    // ---- adaptive sampling + blend
    sample_blend_k<<<(BB * HW + 255) / 256, 256, 0, stream>>>(
        frame1, frame2, out_off1, out_off2, out_wt1, out_wt2,
        out_blend, out_output, BB, HH, WW);
}
